// Attention3_86199993631020
// MI455X (gfx1250) — matrix operand order repaired and checked by reading, not compiled, not run on hardware
//
#include <hip/hip_runtime.h>

#define B_  4
#define M_  4096
#define N_  4096
#define DQ_ 256
#define H_  64

typedef __attribute__((ext_vector_type(16))) __bf16 v16bf;
typedef __attribute__((ext_vector_type(8)))  float  v8f;

__device__ __forceinline__ v8f wmma_bf16(v16bf a, v16bf b, v8f c) {
  // D = A(16x32 bf16) * B(32x16 bf16) + C(16x16 f32)
  return __builtin_amdgcn_wmma_f32_16x16x32_bf16(false, a, false, b, (short)0, c,
                                                 false, false);
}

// Async DMA: global (16B/lane) -> LDS, tracked by ASYNCcnt.
// LDS address operand = low 32 bits of the generic pointer (flat-aperture rule).
__device__ __forceinline__ void async_load_b128(const __bf16* lds_dst,
                                                const __bf16* gsrc) {
  unsigned int l = (unsigned int)(unsigned long long)lds_dst;
  asm volatile("global_load_async_to_lds_b128 %0, %1, off"
               :: "v"(l), "v"(gsrc) : "memory");
}
__device__ __forceinline__ void wait_async0() {
  asm volatile("s_wait_asynccnt 0x0" ::: "memory");
}

// Build an A-fragment (16x32 bf16) for this lane from a row-major source.
// rowk points at (row m = lane&15, col kbase). VGPR v holds
// K = 16*half + 2*v .. +1  (pairs are dword-contiguous).
__device__ __forceinline__ v16bf load_a_frag(const __bf16* rowk, int hf) {
  union { v16bf v; unsigned int u[8]; } r;
#pragma unroll
  for (int vv = 0; vv < 8; ++vv) {
    const int off = hf * 16 + vv * 2;
    r.u[vv] = *(const unsigned int*)(rowk + off);
  }
  return r.v;
}

__device__ __forceinline__ void pack4_store(__bf16* dst, float4 f) {
  union { __bf16 h[4]; uint2 u; } p;
  p.h[0] = (__bf16)f.x; p.h[1] = (__bf16)f.y;
  p.h[2] = (__bf16)f.z; p.h[3] = (__bf16)f.w;
  *(uint2*)dst = p.u;
}

// ---------------------------------------------------------------------------
// Projection GEMM: acc[m,h] = sum_d A[m,d] * W[d,h]
// A fp32 [Mtot,256], W fp32 [256,NH*16].
// SPLIT==false: out0[m][h] = acc (bf16, row-major, Hout wide)        (q path)
// SPLIT==true : h<64 -> out0 = K [B*N,64] row-major
//               h>=64 -> out1 = V^T [B][64][N]                      (kv path)
// ---------------------------------------------------------------------------
template <int NH, bool SPLIT>
__global__ __launch_bounds__(256) void proj_kernel(const float* __restrict__ A,
                                                   const float* __restrict__ W,
                                                   __bf16* __restrict__ out0,
                                                   __bf16* __restrict__ out1) {
  constexpr int Hout = NH * 16;
  __shared__ alignas(32) __bf16 sA[128 * 64];      // A tile, row-major
  __shared__ alignas(32) __bf16 sWt[Hout * 64];    // W tile, transposed [h][d]
  const int tid  = threadIdx.x;
  const int wave = tid >> 5, lane = tid & 31;
  const int hf   = lane >> 4, lm = lane & 15;
  const int m0   = blockIdx.x * 128;

  v8f acc[NH];
#pragma unroll
  for (int i = 0; i < NH; ++i) {
    v8f z = {0.f, 0.f, 0.f, 0.f, 0.f, 0.f, 0.f, 0.f};
    acc[i] = z;
  }

  for (int d0 = 0; d0 < DQ_; d0 += 64) {
    __syncthreads();
    // Stage A tile [128][64]: float4 loads -> packed bf16x4 stores
#pragma unroll
    for (int i = 0; i < 8; ++i) {
      int idx4 = tid + i * 256;               // 2048 float4 total
      int rr = idx4 >> 4, c4 = (idx4 & 15) * 4;
      float4 f = *(const float4*)(A + (size_t)(m0 + rr) * DQ_ + d0 + c4);
      pack4_store(sA + rr * 64 + c4, f);
    }
    // Stage W chunk transposed: sWt[h][dd] = W[d0+dd][h]
    constexpr int WI = (64 * Hout) / (4 * 256);
#pragma unroll
    for (int i = 0; i < WI; ++i) {
      int idx4 = tid + i * 256;
      int dd = (idx4 * 4) / Hout, h4 = (idx4 * 4) % Hout;
      float4 f = *(const float4*)(W + (size_t)(d0 + dd) * Hout + h4);
      sWt[(h4 + 0) * 64 + dd] = (__bf16)f.x;
      sWt[(h4 + 1) * 64 + dd] = (__bf16)f.y;
      sWt[(h4 + 2) * 64 + dd] = (__bf16)f.z;
      sWt[(h4 + 3) * 64 + dd] = (__bf16)f.w;
    }
    __syncthreads();
#pragma unroll
    for (int kc = 0; kc < 64; kc += 32) {
      v16bf a = load_a_frag(sA + (wave * 16 + lm) * 64 + kc, hf);
#pragma unroll
      for (int ht = 0; ht < NH; ++ht) {
        v16bf b = *(const v16bf*)(sWt + (ht * 16 + lm) * 64 + kc + hf * 16);
        acc[ht] = wmma_bf16(a, b, acc[ht]);
      }
    }
  }
  // Epilogue. C layout: element r -> row m = r + 8*hf, col = ht*16+lm
#pragma unroll
  for (int ht = 0; ht < NH; ++ht)
#pragma unroll
    for (int r = 0; r < 8; ++r) {
      const int row = m0 + wave * 16 + r + 8 * hf;  // flattened (b*rows + n)
      const int h   = ht * 16 + lm;
      const __bf16 val = (__bf16)acc[ht][r];
      if constexpr (!SPLIT) {
        out0[(size_t)row * Hout + h] = val;
      } else {
        if (ht < 4) {
          out0[(size_t)row * 64 + h] = val;               // K row-major
        } else {
          const int bb = row / N_, n = row - bb * N_;
          const int d  = h - 64;
          out1[((size_t)bb * 64 + d) * N_ + n] = val;     // V transposed
        }
      }
    }
}

// ---------------------------------------------------------------------------
// Flash attention: q bf16 [B,M,64], k bf16 [B,N,64], vT bf16 [B,64,N],
// out fp32 [B,M,64]. Block = 8 waves = 128 query rows; N in 64-wide tiles.
// K/V tiles double-buffered in LDS and fetched with async global->LDS DMA
// (ASYNCcnt), overlapping the fetch with WMMA + softmax of the current tile.
// ---------------------------------------------------------------------------
__global__ __launch_bounds__(256) void attn_kernel(const __bf16* __restrict__ q,
                                                   const __bf16* __restrict__ k,
                                                   const __bf16* __restrict__ vT,
                                                   float* __restrict__ out) {
  __shared__ alignas(32) __bf16 sK[2][64 * 64];     // [n][d] row-major
  __shared__ alignas(32) __bf16 sVt[2][64 * 64];    // [d][n] (pre-transposed)
  __shared__ alignas(32) __bf16 sP[8 * 16 * 64];    // per-wave P tile [16][64]
  const int tid  = threadIdx.x;
  const int wave = tid >> 5, lane = tid & 31;
  const int hf   = lane >> 4, lm = lane & 15;
  const int tiles_per_b = M_ / 128;
  const int b  = blockIdx.x / tiles_per_b;
  const int m0 = (blockIdx.x % tiles_per_b) * 128;
  const float scale = 0.125f; // 1/sqrt(64)

  // Q fragments for this wave's 16 rows (registers for the whole pass)
  const __bf16* qrow = q + ((size_t)b * M_ + m0 + wave * 16 + lm) * H_;
  const v16bf qa0 = load_a_frag(qrow + 0, hf);
  const v16bf qa1 = load_a_frag(qrow + 32, hf);

  v8f O[4];
  float mi[8], li[8];
#pragma unroll
  for (int dt = 0; dt < 4; ++dt) {
    v8f z = {0.f, 0.f, 0.f, 0.f, 0.f, 0.f, 0.f, 0.f};
    O[dt] = z;
  }
#pragma unroll
  for (int r = 0; r < 8; ++r) { mi[r] = -1.0e30f; li[r] = 0.0f; }

  // Per-thread copy slots: 2 x 16B each for K and Vt (512 x uint4 = 8 KB/tile)
  const int srr  = tid >> 3,          sc8 = (tid & 7) * 8;
  const int srr1 = (tid + 256) >> 3;  // second slot, same sc8
  const __bf16* kb  = k  + (size_t)b * N_ * 64;
  const __bf16* vtb = vT + (size_t)b * 64 * N_;

  auto issue_tile = [&](int nb, int buf) {
    async_load_b128(&sK[buf][srr  * 64 + sc8], kb + (size_t)(nb + srr ) * 64 + sc8);
    async_load_b128(&sK[buf][srr1 * 64 + sc8], kb + (size_t)(nb + srr1) * 64 + sc8);
    async_load_b128(&sVt[buf][srr  * 64 + sc8], vtb + (size_t)srr  * N_ + nb + sc8);
    async_load_b128(&sVt[buf][srr1 * 64 + sc8], vtb + (size_t)srr1 * N_ + nb + sc8);
  };

  issue_tile(0, 0);

  for (int it = 0; it < N_ / 64; ++it) {
    const int buf = it & 1;
    wait_async0();        // this thread's tile-`it` DMA done
    __syncthreads();      // everyone's tile-`it` DMA visible; prev compute done
    if (it + 1 < N_ / 64) issue_tile((it + 1) * 64, buf ^ 1);

    const __bf16* sKb  = sK[buf];
    const __bf16* sVtb = sVt[buf];

    // S = Q * K^T  (4 n-tiles x 2 k-chunks = 8 WMMAs)
    v8f S[4];
#pragma unroll
    for (int nt = 0; nt < 4; ++nt) {
      v8f z = {0.f, 0.f, 0.f, 0.f, 0.f, 0.f, 0.f, 0.f};
      v16bf b0 = *(const v16bf*)(sKb + (nt * 16 + lm) * 64 + 0 + hf * 16);
      z = wmma_bf16(qa0, b0, z);
      v16bf b1 = *(const v16bf*)(sKb + (nt * 16 + lm) * 64 + 32 + hf * 16);
      z = wmma_bf16(qa1, b1, z);
      S[nt] = z;
    }

    // Online softmax: row max across 4 tiles, then across the 16-lane group
    float mnew[8];
#pragma unroll
    for (int r = 0; r < 8; ++r) {
      float v = fmaxf(fmaxf(S[0][r], S[1][r]), fmaxf(S[2][r], S[3][r]));
      mnew[r] = v * scale;
    }
#pragma unroll
    for (int mask = 1; mask < 16; mask <<= 1)
#pragma unroll
      for (int r = 0; r < 8; ++r)
        mnew[r] = fmaxf(mnew[r], __shfl_xor(mnew[r], mask, 32));

#pragma unroll
    for (int r = 0; r < 8; ++r) {
      float mc    = fmaxf(mi[r], mnew[r]);
      float alpha = __expf(mi[r] - mc);
      mi[r] = mc;
      li[r] *= alpha;
#pragma unroll
      for (int dt = 0; dt < 4; ++dt) O[dt][r] *= alpha;
    }

    // P = exp(S*scale - m); stage bf16 into this wave's LDS region
    __bf16* pb = sP + wave * 16 * 64;
#pragma unroll
    for (int nt = 0; nt < 4; ++nt)
#pragma unroll
      for (int r = 0; r < 8; ++r) {
        float p = __expf(S[nt][r] * scale - mi[r]);
        li[r] += p;
        pb[(r + 8 * hf) * 64 + nt * 16 + lm] = (__bf16)p;
      }

    // O += P * V   (2 n-chunks x 4 d-tiles = 8 WMMAs)
#pragma unroll
    for (int nc = 0; nc < 64; nc += 32) {
      v16bf pa = load_a_frag(pb + lm * 64 + nc, hf);
#pragma unroll
      for (int dt = 0; dt < 4; ++dt) {
        v16bf vb = *(const v16bf*)(sVtb + (dt * 16 + lm) * 64 + nc + hf * 16);
        O[dt] = wmma_bf16(pa, vb, O[dt]);
      }
    }
  }

  // Final row-sum reduction and normalize
#pragma unroll
  for (int mask = 1; mask < 16; mask <<= 1)
#pragma unroll
    for (int r = 0; r < 8; ++r)
      li[r] += __shfl_xor(li[r], mask, 32);
#pragma unroll
  for (int r = 0; r < 8; ++r) {
    float inv = 1.0f / li[r];
    size_t orow = ((size_t)b * M_ + m0 + wave * 16 + r + 8 * hf) * H_;
#pragma unroll
    for (int dt = 0; dt < 4; ++dt)
      out[orow + dt * 16 + lm] = O[dt][r] * inv;
  }
}

extern "C" void kernel_launch(void* const* d_in, const int* in_sizes, int n_in,
                              void* d_out, int out_size, void* d_ws, size_t ws_size,
                              hipStream_t stream) {
  const float* x    = (const float*)d_in[0];   // [B, M, 256]
  const float* cond = (const float*)d_in[1];   // [B, N, 256]
  const float* Wq   = (const float*)d_in[2];   // [256, 64]
  const float* Wkv  = (const float*)d_in[3];   // [256, 128]

  __bf16* qws  = (__bf16*)d_ws;                                      // 2 MB
  __bf16* kws  = (__bf16*)((char*)d_ws + (size_t)2 * 1024 * 1024);   // 2 MB
  __bf16* vtws = (__bf16*)((char*)d_ws + (size_t)4 * 1024 * 1024);   // 2 MB

  proj_kernel<4, false><<<(B_ * M_) / 128, 256, 0, stream>>>(x, Wq, qws, nullptr);
  proj_kernel<8, true ><<<(B_ * N_) / 128, 256, 0, stream>>>(cond, Wkv, kws, vtws);
  attn_kernel<<<(B_ * M_) / 128, 256, 0, stream>>>(qws, kws, vtws, (float*)d_out);
}
